// MultiHeadAttention_31224412242379
// MI455X (gfx1250) — matrix operand order repaired and checked by reading, compiled, not run on hardware
//
// MI455X (gfx1250) Music-Transformer MHA, bf16 WMMA throughout.
// ~55 GFLOP of matrix work on v_wmma_f32_16x16x32_bf16 (fp32 accum),
// flash-style attention (no S×S score tensor), skew-trick rel-PE
// (tiny 64x257 GEMM + edge-padded gather instead of an S×S×d_k einsum).
// NOTE: reference pads row_padded with w_k[-1] on BOTH ends, so the rel
// index mapping is (r<0 || r>256) -> 256, r otherwise.
#include <hip/hip_runtime.h>
#include <hip/hip_bf16.h>
#include <math.h>

using bf16 = __bf16;
typedef __attribute__((ext_vector_type(16))) __bf16 v16bf;
typedef __attribute__((ext_vector_type(8)))  __bf16 v8bf;
typedef __attribute__((ext_vector_type(8)))  float  v8f;
typedef __attribute__((ext_vector_type(4)))  float  v4f;

#define DEV __device__ __forceinline__

constexpr int S_LEN = 1024;
constexpr int D_MODEL = 1024;
constexpr int HEADS = 16;
constexpr int D_K = 64;
constexpr int BATCH = 4;
constexpr int M_ROWS = BATCH * S_LEN;     // 4096
constexpr int R_PAD = 272;                // 257 rel columns padded to 17*16

// ---------------- WMMA helpers ----------------

DEV v8f wmma_bf16(v16bf a, v16bf b, v8f c) {
  // D = A(16x32) * B(32x16) + C, fp32 accumulate
  return __builtin_amdgcn_wmma_f32_16x16x32_bf16(false, a, false, b, (short)0, c, false, false);
}

DEV v8f vzero8() {
  v8f c;
#pragma unroll
  for (int i = 0; i < 8; ++i) c[i] = 0.0f;
  return c;
}

// A-matrix 16x32 bf16 fragment (ISA 7.12.2): lane l holds row m=l&15;
// half h=l>>4 selects K groups {h*8..h*8+7} and {16+h*8..16+h*8+7}.
DEV v16bf load_a_frag(const bf16* __restrict__ A, int lda, int m0, int k0, int lane) {
  const int m = lane & 15, half = lane >> 4;
  const bf16* p = A + (size_t)(m0 + m) * lda + k0 + half * 8;
  v8bf lo = *(const v8bf*)p;          // K = k0 + half*8 + 0..7
  v8bf hi = *(const v8bf*)(p + 16);   // K = k0 + 16 + half*8 + 0..7
  v16bf a;
#pragma unroll
  for (int i = 0; i < 8; ++i) { a[i] = lo[i]; a[8 + i] = hi[i]; }
  return a;
}

// Same fragment but converting fp32 source -> bf16 on the fly.
DEV v16bf load_a_frag(const float* __restrict__ A, int lda, int m0, int k0, int lane) {
  const int m = lane & 15, half = lane >> 4;
  const float* p = A + (size_t)(m0 + m) * lda + k0 + half * 8;
  v4f f0 = *(const v4f*)(p);
  v4f f1 = *(const v4f*)(p + 4);
  v4f f2 = *(const v4f*)(p + 16);
  v4f f3 = *(const v4f*)(p + 20);
  v16bf a;
#pragma unroll
  for (int i = 0; i < 4; ++i) {
    a[i]      = (bf16)f0[i];
    a[4 + i]  = (bf16)f1[i];
    a[8 + i]  = (bf16)f2[i];
    a[12 + i] = (bf16)f3[i];
  }
  return a;
}

// B-matrix 32x16 bf16 fragment: lane l holds column n=l&15; half kh=l>>4
// holds K = kh*8 + 0..7 and 16 + kh*8 + 0..7.  B is stored "N-major, K-contiguous"
// which is exactly W's (N,K) row-major layout for x@W.T.
DEV v16bf load_b_frag(const bf16* __restrict__ B, int ldb, int n0, int k0, int lane) {
  const int n = lane & 15, kh = lane >> 4;
  const bf16* p = B + (size_t)(n0 + n) * ldb + k0 + kh * 8;
  v8bf lo = *(const v8bf*)p;
  v8bf hi = *(const v8bf*)(p + 16);
  v16bf b;
#pragma unroll
  for (int i = 0; i < 8; ++i) { b[i] = lo[i]; b[8 + i] = hi[i]; }
  return b;
}

// 16-lane (half-wave) row reductions for the C/D layout (row spread over
// 16 lanes within one VGPR; xor masks 1,2,4,8 stay inside each half).
DEV float red_max16(float v) {
  v = fmaxf(v, __shfl_xor(v, 1, 32));
  v = fmaxf(v, __shfl_xor(v, 2, 32));
  v = fmaxf(v, __shfl_xor(v, 4, 32));
  v = fmaxf(v, __shfl_xor(v, 8, 32));
  return v;
}
DEV float red_sum16(float v) {
  v += __shfl_xor(v, 1, 32);
  v += __shfl_xor(v, 2, 32);
  v += __shfl_xor(v, 4, 32);
  v += __shfl_xor(v, 8, 32);
  return v;
}

// ---------------- Kernel 0a: fp32 -> bf16 weight conversion ----------------

__global__ __launch_bounds__(256) void cvt_bf16_kernel(const float* __restrict__ src,
                                                       bf16* __restrict__ dst, int n) {
  int i = blockIdx.x * 256 + threadIdx.x;
  if (i < n) dst[i] = (bf16)src[i];
}

// ---------------- Kernel 0b: pad + convert w_k (257x64 -> 272x64 bf16) ----------------

__global__ __launch_bounds__(256) void prep_wk_kernel(const float* __restrict__ wk,
                                                      bf16* __restrict__ wkbf) {
  int i = blockIdx.x * 256 + threadIdx.x;
  if (i < R_PAD * D_K) {
    int r = i >> 6, d = i & 63;
    int rs = r > 256 ? 256 : r;   // replicate last row into the pad
    wkbf[i] = (bf16)wk[rs * D_K + d];
  }
}

// ---------------- Kernels 1-3/6: C = A @ W.T + bias  (M=4096, N=K=1024) ----------------
// 8 waves/block (2x4), each wave owns a 32x64 strip (2x4 WMMA tiles -> 8 WMMA
// per 6 fragment loads per k-step); block tile = 64x256.  W pre-converted to
// bf16.  Epilogue optionally writes: bf16 row-major (for attention), bf16
// transposed V^T[(b,h),d,s] (for P@V B-frags), and/or fp32 (final output).

template <typename AT>
__global__ __launch_bounds__(256) void gemm_bias_kernel(
    const AT* __restrict__ A, const bf16* __restrict__ Wb, const float* __restrict__ bias,
    bf16* __restrict__ outBf, bf16* __restrict__ outVt, float* __restrict__ outF) {
  const int lane = threadIdx.x & 31, wid = threadIdx.x >> 5;
  const int m0 = blockIdx.y * 64 + (wid >> 2) * 32;
  const int n0 = blockIdx.x * 256 + (wid & 3) * 64;

  v8f c[2][4];
#pragma unroll
  for (int i = 0; i < 2; ++i)
#pragma unroll
    for (int j = 0; j < 4; ++j) c[i][j] = vzero8();

#pragma unroll 2
  for (int kk = 0; kk < D_MODEL; kk += 32) {
    v16bf a0 = load_a_frag(A, D_MODEL, m0, kk, lane);
    v16bf a1 = load_a_frag(A, D_MODEL, m0 + 16, kk, lane);
#pragma unroll
    for (int j = 0; j < 4; ++j) {
      v16bf b = load_b_frag(Wb, D_MODEL, n0 + j * 16, kk, lane);
      c[0][j] = wmma_bf16(a0, b, c[0][j]);
      c[1][j] = wmma_bf16(a1, b, c[1][j]);
    }
  }

  const int nl = lane & 15, half = lane >> 4;
#pragma unroll
  for (int i = 0; i < 2; ++i) {
#pragma unroll
    for (int j = 0; j < 4; ++j) {
      const int nn = n0 + j * 16 + nl;
      const float bv = bias[nn];
#pragma unroll
      for (int r = 0; r < 8; ++r) {
        const int mm = m0 + i * 16 + r + half * 8;
        const float val = c[i][j][r] + bv;
        if (outBf) outBf[(size_t)mm * D_MODEL + nn] = (bf16)val;
        if (outVt) {
          const int b_ = mm >> 10, s_ = mm & 1023, h_ = nn >> 6, d_ = nn & 63;
          outVt[(((size_t)b_ * HEADS + h_) * D_K + d_) * S_LEN + s_] = (bf16)val;
        }
        if (outF) outF[(size_t)mm * D_MODEL + nn] = val;
      }
    }
  }
}

// ---------------- Kernel 4: QW[h][m][r] = Q_h[m] . w_k[r]   (K = 64) ----------------

__global__ __launch_bounds__(256) void qw_kernel(const bf16* __restrict__ Qbf,
                                                 const bf16* __restrict__ wkbf,
                                                 float* __restrict__ QW) {
  const int lane = threadIdx.x & 31, wid = threadIdx.x >> 5;
  const int h = blockIdx.z;
  const int m0 = (blockIdx.y * 8 + wid) * 16;
  const int n0 = blockIdx.x * 16;          // rel-index tile, grid.x = 17
  v8f c = vzero8();
#pragma unroll
  for (int kk = 0; kk < D_K; kk += 32) {
    v16bf a = load_a_frag(Qbf, D_MODEL, m0, h * D_K + kk, lane);
    v16bf b = load_b_frag(wkbf, D_K, n0, kk, lane);
    c = wmma_bf16(a, b, c);
  }
  const int nl = lane & 15, half = lane >> 4;
#pragma unroll
  for (int r = 0; r < 8; ++r)
    QW[((size_t)h * M_ROWS + m0 + r + half * 8) * R_PAD + n0 + nl] = c[r];
}

// ---------------- Kernel 5: flash attention with rel-PE + mask ----------------
// One (b,h) per blockIdx.y; 8 waves x 16 queries = 128 queries per block.
// Key tiles of 32 so the softmax P tile is exactly one 16x32 bf16 A-fragment.

__global__ __launch_bounds__(256) void flash_kernel(
    const bf16* __restrict__ Qbf, const bf16* __restrict__ Kbf, const bf16* __restrict__ Vt,
    const float* __restrict__ QW, const int* __restrict__ mask, bf16* __restrict__ ctx) {
  __shared__ __align__(16) bf16 ldsP[8][16 * 32];

  const int lane = threadIdx.x & 31, wid = threadIdx.x >> 5;
  const int bh = blockIdx.y, b = bh >> 4, h = bh & 15;
  const int q0 = blockIdx.x * 128 + wid * 16;          // query s base for this wave
  const int nl = lane & 15, half = lane >> 4;

  // Q fragments for d_k = 64 (two 16x32 A-frags), reused every iteration.
  const v16bf aq0 = load_a_frag(Qbf, D_MODEL, b * S_LEN + q0, h * D_K, lane);
  const v16bf aq1 = load_a_frag(Qbf, D_MODEL, b * S_LEN + q0, h * D_K + 32, lane);

  v8f o0 = vzero8(), o1 = vzero8(), o2 = vzero8(), o3 = vzero8();
  float mx[8], sm[8];
#pragma unroll
  for (int r = 0; r < 8; ++r) { mx[r] = -INFINITY; sm[r] = 0.0f; }

  const float scale = 0.125f;  // 1/sqrt(64)

  for (int kb = 0; kb < S_LEN; kb += 32) {
    // Prefetch next key tile (gfx1250 global_prefetch_b8); one row per lane.
    if (kb + 32 < S_LEN) {
      __builtin_prefetch(Kbf + (size_t)(b * S_LEN + kb + 32 + lane) * D_MODEL + h * D_K, 0, 3);
      __builtin_prefetch(Vt + (size_t)(bh * D_K + lane) * S_LEN + kb + 32, 0, 3);
      __builtin_prefetch(Vt + (size_t)(bh * D_K + 32 + lane) * S_LEN + kb + 32, 0, 3);
      __builtin_prefetch(mask + ((size_t)b * S_LEN + q0 + nl) * S_LEN + kb + 32 + half * 16, 0, 3);
    }
    // Seed C with the relative-PE term via the skew-trick gather.
    // Reference pads with w_k[-1] on BOTH ends: out-of-window -> index 256.
    v8f c0, c1;
#pragma unroll
    for (int r = 0; r < 8; ++r) {
      const int i_s = q0 + r + half * 8;
      const size_t qrow = ((size_t)h * M_ROWS + b * S_LEN + i_s) * R_PAD;
      const int rr0 = kb + nl - i_s + 128;
      const int rr1 = kb + 16 + nl - i_s + 128;
      c0[r] = QW[qrow + ((unsigned)rr0 > 256u ? 256 : rr0)];
      c1[r] = QW[qrow + ((unsigned)rr1 > 256u ? 256 : rr1)];
    }
    // scores += Q . K^T  (B-frags = K rows, d-contiguous)
    {
      v16bf bk;
      bk = load_b_frag(Kbf, D_MODEL, b * S_LEN + kb,      h * D_K,      lane); c0 = wmma_bf16(aq0, bk, c0);
      bk = load_b_frag(Kbf, D_MODEL, b * S_LEN + kb,      h * D_K + 32, lane); c0 = wmma_bf16(aq1, bk, c0);
      bk = load_b_frag(Kbf, D_MODEL, b * S_LEN + kb + 16, h * D_K,      lane); c1 = wmma_bf16(aq0, bk, c1);
      bk = load_b_frag(Kbf, D_MODEL, b * S_LEN + kb + 16, h * D_K + 32, lane); c1 = wmma_bf16(aq1, bk, c1);
    }
    // scale + mask (-1e9, matching the reference's finite fill)
#pragma unroll
    for (int r = 0; r < 8; ++r) {
      const int i_s = q0 + r + half * 8;
      const int* mrow = mask + ((size_t)b * S_LEN + i_s) * S_LEN + kb;
      c0[r] = (mrow[nl]      == 0) ? -1e9f : c0[r] * scale;
      c1[r] = (mrow[16 + nl] == 0) ? -1e9f : c1[r] * scale;
    }
    // Online softmax (row stats live replicated across each 16-lane half).
    float p0[8], p1[8];
#pragma unroll
    for (int r = 0; r < 8; ++r) {
      const float tm = fmaxf(red_max16(c0[r]), red_max16(c1[r]));
      const float nm = fmaxf(mx[r], tm);
      const float corr = __expf(mx[r] - nm);
      p0[r] = __expf(c0[r] - nm);
      p1[r] = __expf(c1[r] - nm);
      const float rs = red_sum16(p0[r]) + red_sum16(p1[r]);
      sm[r] = sm[r] * corr + rs;
      mx[r] = nm;
      o0[r] *= corr; o1[r] *= corr; o2[r] *= corr; o3[r] *= corr;
    }
    // P (C layout) -> LDS -> 16x32 A-fragment. Same-wave DS ops are in-order.
    bf16* pl = ldsP[wid];
#pragma unroll
    for (int r = 0; r < 8; ++r) {
      const int row = r + half * 8;
      pl[row * 32 + nl]      = (bf16)p0[r];
      pl[row * 32 + 16 + nl] = (bf16)p1[r];
    }
    const v16bf pf = load_a_frag(pl, 32, 0, 0, lane);
    // O += P @ V   (B-frags straight from global V^T, s-contiguous)
    o0 = wmma_bf16(pf, load_b_frag(Vt, S_LEN, bh * D_K +  0, kb, lane), o0);
    o1 = wmma_bf16(pf, load_b_frag(Vt, S_LEN, bh * D_K + 16, kb, lane), o1);
    o2 = wmma_bf16(pf, load_b_frag(Vt, S_LEN, bh * D_K + 32, kb, lane), o2);
    o3 = wmma_bf16(pf, load_b_frag(Vt, S_LEN, bh * D_K + 48, kb, lane), o3);
  }

  // Normalize and emit bf16 context in concat layout [b*S+s][h*64+d].
#pragma unroll
  for (int r = 0; r < 8; ++r) {
    const float inv = 1.0f / sm[r];
    const int i_s = q0 + r + half * 8;
    const size_t rowoff = ((size_t)b * S_LEN + i_s) * D_MODEL + h * D_K;
    ctx[rowoff +  0 + nl] = (bf16)(o0[r] * inv);
    ctx[rowoff + 16 + nl] = (bf16)(o1[r] * inv);
    ctx[rowoff + 32 + nl] = (bf16)(o2[r] * inv);
    ctx[rowoff + 48 + nl] = (bf16)(o3[r] * inv);
  }
}

// ---------------- Host-side orchestration ----------------

extern "C" void kernel_launch(void* const* d_in, const int* in_sizes, int n_in,
                              void* d_out, int out_size, void* d_ws, size_t ws_size,
                              hipStream_t stream) {
  (void)in_sizes; (void)n_in; (void)out_size; (void)ws_size;
  const float* q    = (const float*)d_in[0];
  const float* k    = (const float*)d_in[1];
  const float* v    = (const float*)d_in[2];
  const int*   mask = (const int*)d_in[3];
  const float* Wq   = (const float*)d_in[4];
  const float* bq   = (const float*)d_in[5];
  const float* Wk   = (const float*)d_in[6];
  const float* bk   = (const float*)d_in[7];
  const float* Wv   = (const float*)d_in[8];
  const float* bv   = (const float*)d_in[9];
  const float* Wo   = (const float*)d_in[10];
  const float* bo   = (const float*)d_in[11];
  const float* w_k  = (const float*)d_in[12];
  float* out = (float*)d_out;

  // Workspace layout (~113 MB)
  char* ws = (char*)d_ws;
  constexpr size_t SZ_ACT = (size_t)M_ROWS * D_MODEL * sizeof(bf16);   // 8 MB
  constexpr size_t SZ_W   = (size_t)D_MODEL * D_MODEL * sizeof(bf16);  // 2 MB
  bf16*  Qbf  = (bf16*)(ws + 0 * SZ_ACT);
  bf16*  Kbf  = (bf16*)(ws + 1 * SZ_ACT);
  bf16*  Vt   = (bf16*)(ws + 2 * SZ_ACT);   // [(b,h), d, s]
  bf16*  ctx  = (bf16*)(ws + 3 * SZ_ACT);
  bf16*  Wqb  = (bf16*)(ws + 4 * SZ_ACT);
  bf16*  Wkb  = (bf16*)(ws + 4 * SZ_ACT + 1 * SZ_W);
  bf16*  Wvb  = (bf16*)(ws + 4 * SZ_ACT + 2 * SZ_W);
  bf16*  Wob  = (bf16*)(ws + 4 * SZ_ACT + 3 * SZ_W);
  bf16*  wkbf = (bf16*)(ws + 4 * SZ_ACT + 4 * SZ_W);
  float* QW   = (float*)(ws + 4 * SZ_ACT + 4 * SZ_W + 65536);  // [h][m][272] fp32, ~71 MB

  // Prep: weights -> bf16, padded rel table -> bf16.
  const int nW = D_MODEL * D_MODEL;
  cvt_bf16_kernel<<<(nW + 255) / 256, 256, 0, stream>>>(Wq, Wqb, nW);
  cvt_bf16_kernel<<<(nW + 255) / 256, 256, 0, stream>>>(Wk, Wkb, nW);
  cvt_bf16_kernel<<<(nW + 255) / 256, 256, 0, stream>>>(Wv, Wvb, nW);
  cvt_bf16_kernel<<<(nW + 255) / 256, 256, 0, stream>>>(Wo, Wob, nW);
  prep_wk_kernel<<<(R_PAD * D_K + 255) / 256, 256, 0, stream>>>(w_k, wkbf);

  const dim3 gg(D_MODEL / 256, M_ROWS / 64);  // (4, 64)
  gemm_bias_kernel<float><<<gg, 256, 0, stream>>>(q, Wqb, bq, Qbf, nullptr, nullptr);
  gemm_bias_kernel<float><<<gg, 256, 0, stream>>>(k, Wkb, bk, Kbf, nullptr, nullptr);
  gemm_bias_kernel<float><<<gg, 256, 0, stream>>>(v, Wvb, bv, nullptr, Vt, nullptr);

  qw_kernel<<<dim3(R_PAD / 16, M_ROWS / 128, HEADS), 256, 0, stream>>>(Qbf, wkbf, QW);

  flash_kernel<<<dim3(S_LEN / 128, BATCH * HEADS), 256, 0, stream>>>(Qbf, Kbf, Vt, QW, mask, ctx);

  gemm_bias_kernel<bf16><<<gg, 256, 0, stream>>>(ctx, Wob, bo, nullptr, nullptr, out);
}
